// TIPWrapper_66907000537234
// MI455X (gfx1250) — matrix operand order repaired and checked by reading, not compiled, not run on hardware
//
#include <hip/hip_runtime.h>

typedef __attribute__((ext_vector_type(16))) _Float16 v16h;
typedef __attribute__((ext_vector_type(8)))  _Float16 h8;
typedef __attribute__((ext_vector_type(8)))  float    v8f;
typedef __attribute__((ext_vector_type(4)))  float    f4;
typedef __attribute__((ext_vector_type(4)))  int      v4i;

typedef __attribute__((address_space(1))) v4i* gv4i_p;   // global 16B chunk
typedef __attribute__((address_space(3))) v4i* lv4i_p;   // LDS    16B chunk

#define B_ROWS  2048
#define D_DIM   1024
#define NCACHE  32000
#define C_CLS   1000

#define BM 128      // block tile rows (feats/B)
#define BN 128      // block tile cols (cache entries/N)
#define BK 32       // K step = WMMA K
#define LDA 40      // padded LDS stride in halves (80B: 16B aligned, bank-staggered)
#define LDB 40

#if __has_builtin(__builtin_amdgcn_global_load_async_to_lds_b128)
#define HAS_ASYNC_LDS 1
#else
#define HAS_ASYNC_LDS 0
#endif

// one 16-byte global -> LDS copy (async on CDNA5 if the builtin exists)
__device__ __forceinline__ void copy16(const _Float16* __restrict__ g,
                                       _Float16* l) {
#if HAS_ASYNC_LDS
  __builtin_amdgcn_global_load_async_to_lds_b128(
      (gv4i_p)(void*)g, (lv4i_p)(void*)l, 0, 0);
#else
  *(h8*)l = *(const h8*)g;
#endif
}

__device__ __forceinline__ void wait_stage_done() {
#if HAS_ASYNC_LDS
#if __has_builtin(__builtin_amdgcn_s_wait_asynccnt)
  __builtin_amdgcn_s_wait_asynccnt(0);
#else
  asm volatile("s_wait_asynccnt 0" ::: "memory");
#endif
#endif
  __syncthreads();
}

// ---------------------------------------------------------------------------
// fp32 -> fp16 bulk conversion
// ---------------------------------------------------------------------------
__global__ void cvt_f32_to_f16(const float* __restrict__ src,
                               _Float16* __restrict__ dst, int n) {
  int i = (blockIdx.x * blockDim.x + threadIdx.x) * 4;
  if (i + 3 < n) {
    f4 v = *(const f4*)(src + i);
    dst[i + 0] = (_Float16)v.x;
    dst[i + 1] = (_Float16)v.y;
    dst[i + 2] = (_Float16)v.z;
    dst[i + 3] = (_Float16)v.w;
  }
}

// ---------------------------------------------------------------------------
// Fused: affinity GEMM (WMMA f16->f32) + exp + alpha-scaled scatter into out.
// out must already contain tv_logits.  Double-buffered LDS, async staging.
// ---------------------------------------------------------------------------
__global__ __launch_bounds__(256)
void tip_fused(const _Float16* __restrict__ A16,   // [B_ROWS][D_DIM]
               const _Float16* __restrict__ W16,   // [NCACHE][D_DIM]
               const float*    __restrict__ beta_alpha,
               const int*      __restrict__ labels,
               float*          __restrict__ out)   // [B_ROWS][C_CLS]
{
  __shared__ _Float16 As[2][BM * LDA];
  __shared__ _Float16 Ws[2][BN * LDB];

  const int tid   = threadIdx.x;
  const int lane  = tid & 31;
  const int wave  = tid >> 5;
  const int wm    = wave & 3;    // 4 wave-rows: 32 M each
  const int wn    = wave >> 2;   // 2 wave-cols: 64 N each
  const int khalf = lane >> 4;   // 0: lanes 0-15, 1: lanes 16-31
  const int l16   = lane & 15;

  const int bm0 = blockIdx.y * BM;
  const int bn0 = blockIdx.x * BN;

  const float beta  = beta_alpha[0];
  const float alpha = beta_alpha[1];

  // per-thread staging chunk coordinates (2 chunks each for A and W):
  // A tile 128x32 halves = 512 x 16B chunks; same for W tile.
  const int rowc = tid >> 2;            // chunk row for ch = tid
  const int colc = (tid & 3) * 8;       // chunk col (halves)

  v8f acc[2][4] = {};   // 2 (M) x 4 (N) 16x16 fp32 accumulators

  // ---- prologue: stage K-slab 0 into buffer 0 ----
  {
    copy16(&A16[(size_t)(bm0 + rowc) * D_DIM + colc],        &As[0][rowc * LDA + colc]);
    copy16(&A16[(size_t)(bm0 + 64 + rowc) * D_DIM + colc],   &As[0][(64 + rowc) * LDA + colc]);
    copy16(&W16[(size_t)(bn0 + rowc) * D_DIM + colc],        &Ws[0][rowc * LDB + colc]);
    copy16(&W16[(size_t)(bn0 + 64 + rowc) * D_DIM + colc],   &Ws[0][(64 + rowc) * LDB + colc]);
  }
  wait_stage_done();

  int buf = 0;
  for (int k0 = 0; k0 < D_DIM; k0 += BK) {
    // ---- stage next K-slab into the other buffer (overlaps with WMMAs) ----
    if (k0 + BK < D_DIM) {
      const int kn = k0 + BK;
      copy16(&A16[(size_t)(bm0 + rowc) * D_DIM + kn + colc],      &As[buf ^ 1][rowc * LDA + colc]);
      copy16(&A16[(size_t)(bm0 + 64 + rowc) * D_DIM + kn + colc], &As[buf ^ 1][(64 + rowc) * LDA + colc]);
      copy16(&W16[(size_t)(bn0 + rowc) * D_DIM + kn + colc],      &Ws[buf ^ 1][rowc * LDB + colc]);
      copy16(&W16[(size_t)(bn0 + 64 + rowc) * D_DIM + kn + colc], &Ws[buf ^ 1][(64 + rowc) * LDB + colc]);
    }

    // ---- fragments (ISA 16-bit A 16x32 / B 32x16 lane layouts) ----
    const _Float16* sa = As[buf];
    const _Float16* sw = Ws[buf];
    v16h afrag[2], bfrag[4];
    #pragma unroll
    for (int mt = 0; mt < 2; ++mt) {
      int arow = (wm * 32 + mt * 16 + l16) * LDA;
      h8 lo = *(const h8*)(&sa[arow + khalf * 8]);        // K = kh*8 .. +7
      h8 hi = *(const h8*)(&sa[arow + 16 + khalf * 8]);   // K = 16+kh*8 .. +7
      afrag[mt] = __builtin_shufflevector(lo, hi,
          0,1,2,3,4,5,6,7,8,9,10,11,12,13,14,15);
    }
    #pragma unroll
    for (int nt = 0; nt < 4; ++nt) {
      int brow = (wn * 64 + nt * 16 + l16) * LDB;
      h8 lo = *(const h8*)(&sw[brow + khalf * 8]);        // K = kh*8 .. +7
      h8 hi = *(const h8*)(&sw[brow + 16 + khalf * 8]);   // K = 16+kh*8 .. +7
      bfrag[nt] = __builtin_shufflevector(lo, hi,
          0,1,2,3,4,5,6,7,8,9,10,11,12,13,14,15);
    }

    #pragma unroll
    for (int mt = 0; mt < 2; ++mt)
      #pragma unroll
      for (int nt = 0; nt < 4; ++nt)
        acc[mt][nt] = __builtin_amdgcn_wmma_f32_16x16x32_f16(
            false, afrag[mt], false, bfrag[nt],
            (short)0, acc[mt][nt], false, false);

    // next buffer staged + everyone done reading this one
    wait_stage_done();
    buf ^= 1;
  }

  // ---- epilogue: act = exp(beta*(aff-1)); out[m, label[n]] += alpha*act ----
  #pragma unroll
  for (int nt = 0; nt < 4; ++nt) {
    int n   = bn0 + wn * 64 + nt * 16 + l16;   // C/D layout: N = lane%16
    int lbl = labels[n];
    #pragma unroll
    for (int mt = 0; mt < 2; ++mt) {
      int mbase = bm0 + wm * 32 + mt * 16 + khalf * 8;  // M = r + 8*(lane/16)
      #pragma unroll
      for (int r = 0; r < 8; ++r) {
        float aff = acc[mt][nt][r];
        float val = alpha * __expf(beta * (aff - 1.0f));
        atomicAdd(&out[(size_t)(mbase + r) * C_CLS + lbl], val);
      }
    }
  }
}

// ---------------------------------------------------------------------------
extern "C" void kernel_launch(void* const* d_in, const int* in_sizes, int n_in,
                              void* d_out, int out_size, void* d_ws, size_t ws_size,
                              hipStream_t stream) {
  const float* feats  = (const float*)d_in[0];   // [B, D]
  const float* tv     = (const float*)d_in[1];   // [B, C]
  const float* W      = (const float*)d_in[2];   // [N, D]
  const float* ba     = (const float*)d_in[3];   // [2]
  const int*   labels = (const int*)d_in[4];     // [N]
  float* out = (float*)d_out;

  _Float16* A16 = (_Float16*)d_ws;                                   // 4 MB
  _Float16* W16 = (_Float16*)((char*)d_ws +
                  (size_t)B_ROWS * D_DIM * sizeof(_Float16));        // 64 MB

  // out = tv_logits (scatter target base)
  (void)hipMemcpyAsync(out, tv, (size_t)B_ROWS * C_CLS * sizeof(float),
                       hipMemcpyDeviceToDevice, stream);

  int nA = B_ROWS * D_DIM;           // 2,097,152
  int nW = NCACHE * D_DIM;           // 32,768,000
  cvt_f32_to_f16<<<(nA / 4 + 255) / 256, 256, 0, stream>>>(feats, A16, nA);
  cvt_f32_to_f16<<<(nW / 4 + 255) / 256, 256, 0, stream>>>(W, W16, nW);

  dim3 grid(NCACHE / BN, B_ROWS / BM);   // (250, 16)
  tip_fused<<<grid, 256, 0, stream>>>(A16, W16, ba, labels, out);
}
